// GRU_58471684767776
// MI455X (gfx1250) — compile-verified, matrix operand order checked by reading, not run on hardware
//
#include <hip/hip_runtime.h>
#include <math.h>

// GRU: L=512, B=128, D=512, H=1024, K = H+D = 1536
#define LSEQ  512
#define BATCH 128
#define DIN   512
#define HID   1024
#define KTOT  (HID + DIN)      // 1536
#define KSTEPS (KTOT / 32)     // 48 K-steps of 32 (bf16 WMMA K)

typedef __attribute__((ext_vector_type(16))) __bf16 v16bf;
typedef __attribute__((ext_vector_type(8)))  float  v8f;
typedef __attribute__((ext_vector_type(4)))  unsigned int v4u;
typedef __attribute__((ext_vector_type(8)))  int    v8i;
typedef __attribute__((ext_vector_type(4)))  int    v4i;

struct U4x2 { uint4 lo, hi; };
union FragU { U4x2 u; v16bf v; };

#if defined(__has_builtin)
#if __has_builtin(__builtin_amdgcn_tensor_load_to_lds) && __has_builtin(__builtin_amdgcn_s_wait_tensorcnt)
#define USE_TDM 1
#endif
#endif
#ifndef USE_TDM
#define USE_TDM 0
#endif

// round-to-nearest-even fp32 -> bf16
__device__ __forceinline__ unsigned short f2bf(float f) {
  union { float f; unsigned int u; } c; c.f = f;
  unsigned int u = c.u;
  unsigned int r = u + 0x7FFFu + ((u >> 16) & 1u);
  return (unsigned short)(r >> 16);
}

__device__ __forceinline__ float sigm(float v) {
  return 1.0f / (1.0f + __expf(-v));
}

#if USE_TDM
// Issue one TDM 2D tile load: 32 bf16 elements x 128 rows from global (row
// pitch = row_elems bf16) into contiguous LDS at lds_off. D# bit packing per
// CDNA5 ISA (group0: count/lds/global/type,
// group1: data_size=2B, tensor dims, tile dims, dim0 stride). 2D -> groups
// 2/3 zero. Issued per-wave; call from ONE wave only, then s_wait_tensorcnt.
// This toolchain uses the 6-arg builtin form:
//   (uint32x4 g0, int32x8 g1, int32x4 g2, int32x4 g3, int32x8 g4, i32 cpol)
__device__ __forceinline__ void tdm_load_tile(unsigned int lds_off,
                                              const void* gptr,
                                              unsigned int row_elems) {
  unsigned long long ga = (unsigned long long)gptr;
  v4u g0;
  g0[0] = 0x1u;                                    // count=1, user mode
  g0[1] = lds_off;                                 // LDS byte address
  g0[2] = (unsigned int)(ga & 0xFFFFFFFFu);        // global_addr[31:0]
  g0[3] = (unsigned int)((ga >> 32) & 0x01FFFFFFu) // global_addr[56:32]
          | (2u << 30);                            // type=2 ("image")
  v8i g1;
  g1[0] = (1 << 16);        // workgroup_mask=0, data_size=1 (2 bytes)
  g1[1] = (32 << 16);       // tensor_dim0 = 32  (bits 79:48, low half)
  g1[2] = (128 << 16);      // tensor_dim0 hi=0 | tensor_dim1 = 128 (low)
  g1[3] = (32 << 16);       // tensor_dim1 hi=0 | tile_dim0 = 32
  g1[4] = 128;              // tile_dim1 = 128, tile_dim2 = 0
  g1[5] = (int)row_elems;   // tensor_dim0_stride[31:0] (elements)
  g1[6] = 0;                // stride hi / dim1_stride lo
  g1[7] = 0;
  v4i gz4 = {0, 0, 0, 0};
  v8i gz8 = {0, 0, 0, 0, 0, 0, 0, 0};
  __builtin_amdgcn_tensor_load_to_lds(g0, g1, gz4, gz4, gz8, 0);
}
#endif

// ---------------- one-time setup kernels ----------------

__global__ void k_convert_w(const float* __restrict__ Wz, const float* __restrict__ Wr,
                            const float* __restrict__ Wh,
                            unsigned short* __restrict__ Wzbf, unsigned short* __restrict__ Wrbf,
                            unsigned short* __restrict__ Whbf) {
  const size_t n = (size_t)HID * KTOT;
  for (size_t i = (size_t)blockIdx.x * blockDim.x + threadIdx.x; i < n;
       i += (size_t)gridDim.x * blockDim.x) {
    Wzbf[i] = f2bf(Wz[i]);
    Wrbf[i] = f2bf(Wr[i]);
    Whbf[i] = f2bf(Wh[i]);
  }
}

__global__ void k_init_h(const float* __restrict__ h0, float* __restrict__ hbuf,
                         unsigned short* __restrict__ hbf) {
  size_t i = (size_t)blockIdx.x * blockDim.x + threadIdx.x; // B*H total
  float v = h0[i];
  hbuf[i] = v;
  hbf[i]  = f2bf(v);
}

// ---------------- shared GEMM plumbing ----------------
// LDS A-tile: 128 rows x 32 bf16 (64B/row). Each lane's WMMA A fragment is
// two aligned 16B chunks (ds_load_b128) per the 16-bit A layout:
// lanes 0-15 K in [0,8)+[16,24), lanes 16-31 K in [8,16)+[24,32).

// fp32 x_t -> bf16 staging (columns H..K-1 of hx)
__device__ __forceinline__ void stage_x(unsigned int* lds32, int t, int k0,
                                        const float* __restrict__ x) {
  const int tid = threadIdx.x;
  const float* xb = x + ((size_t)t * BATCH) * DIN + (k0 - HID);
  #pragma unroll
  for (int i = 0; i < 8; ++i) {
    int e   = tid + i * 256;         // e = row*16 + kk (uint granularity)
    int row = e >> 4;
    int kk  = e & 15;
    const float* p = xb + (size_t)row * DIN + kk * 2;
    unsigned int lo = f2bf(p[0]);
    unsigned int hi = f2bf(p[1]);
    lds32[e] = lo | (hi << 16);
  }
}

// bf16 source fallback staging (used only if TDM builtin unavailable)
__device__ __forceinline__ void stage_bf(unsigned int* lds32, int k0,
                                         const unsigned int* __restrict__ src32) {
  const int tid = threadIdx.x;
  const int base = k0 >> 1;
  #pragma unroll
  for (int i = 0; i < 8; ++i) {
    int e   = tid + i * 256;
    int row = e >> 4;
    int kk  = e & 15;
    lds32[e] = src32[row * (HID / 2) + base + kk];
  }
}

// Core per-wave GEMM: acc[4] covers rows [mbase,mbase+64) x cols [ncol,ncol+16)
// of C = A[128,K] * W^T, W row-major [H,K] bf16.
__device__ __forceinline__ void wave_gemm(v8f acc[4], uint4* ldsA4, int t,
                                          const float* __restrict__ x,
                                          const unsigned short* __restrict__ srcbf,
                                          const unsigned short* __restrict__ W,
                                          int ncol, int mbase, int lane) {
  unsigned int* lds32 = (unsigned int*)ldsA4;
  const int lrow = lane & 15;
  const int lsel = (lane >> 4) & 1;   // K-chunk select per ISA bf16 layout
  const unsigned short* wrow = W + (size_t)(ncol + lrow) * KTOT;
#if USE_TDM
  const unsigned int lds_off = (unsigned int)(unsigned long long)(void*)ldsA4;
#endif

  for (int ks = 0; ks < KSTEPS; ++ks) {
    const int k0 = ks * 32;
    __syncthreads();                  // previous tile fully consumed
    if (k0 < HID) {
#if USE_TDM
      if (threadIdx.x < 32) {         // wave 0 drives the Tensor Data Mover
        tdm_load_tile(lds_off, srcbf + k0, HID);
        __builtin_amdgcn_s_wait_tensorcnt(0);
      }
#else
      stage_bf(lds32, k0, (const unsigned int*)srcbf);
#endif
    } else {
      stage_x(lds32, t, k0, x);
    }
    __syncthreads();

    // WGP-scope prefetch of next K-step of this lane's weight row
    // (locality 3 -> near cache; weights are L2-resident after step 0)
    if (ks + 1 < KSTEPS) __builtin_prefetch(wrow + k0 + 32, 0, 3);

    FragU bf;
    const uint4* wp = (const uint4*)(wrow + k0);   // 64B-aligned
    bf.u.lo = wp[lsel];
    bf.u.hi = wp[2 + lsel];

    FragU af[4];
    #pragma unroll
    for (int mi = 0; mi < 4; ++mi) {               // preload all A fragments
      const int row = mbase + mi * 16 + lrow;
      af[mi].u.lo = ldsA4[row * 4 + lsel];
      af[mi].u.hi = ldsA4[row * 4 + 2 + lsel];
    }
    #pragma unroll
    for (int mi = 0; mi < 4; ++mi) {
      acc[mi] = __builtin_amdgcn_wmma_f32_16x16x32_bf16(
          false, af[mi].v, false, bf.v, (short)0, acc[mi], false, false);
    }
  }
}

// ---------------- step kernel 1: z and r gates ----------------
// grid = 32 blocks: gate = blockIdx.x>>4 (0=z, 1=r), 64 cols per block.
__global__ void __launch_bounds__(256)
k_gates(int t, const float* __restrict__ x,
        const unsigned short* __restrict__ Wzbf, const unsigned short* __restrict__ Wrbf,
        const float* __restrict__ bz, const float* __restrict__ br,
        const unsigned short* __restrict__ hbf, const float* __restrict__ hbuf,
        float* __restrict__ zbuf, unsigned short* __restrict__ rhbf) {
  __shared__ uint4 ldsA4[128 * 4];                 // 128 x 32 bf16 = 8KB
  const int gate   = blockIdx.x >> 4;
  const int nblock = (blockIdx.x & 15) * 64;
  const int wave   = threadIdx.x >> 5;
  const int lane   = threadIdx.x & 31;
  const int ncol   = nblock + (wave & 3) * 16;
  const int mbase  = (wave >> 2) * 64;

  const unsigned short* W = gate ? Wrbf : Wzbf;
  const float* bias       = gate ? br   : bz;

  v8f acc[4] = {};
  wave_gemm(acc, ldsA4, t, x, hbf, W, ncol, mbase, lane);

  const int lrow = lane & 15;
  const int lsel = (lane >> 4) & 1;
  const int j    = ncol + lrow;
  const float bj = bias[j];
  #pragma unroll
  for (int mi = 0; mi < 4; ++mi) {
    #pragma unroll
    for (int v = 0; v < 8; ++v) {
      const int    brow = mbase + mi * 16 + v + lsel * 8;
      const size_t idx  = (size_t)brow * HID + j;
      const float  g    = sigm(acc[mi][v] + bj);
      if (gate == 0) {
        zbuf[idx] = g;                              // update gate
      } else {
        rhbf[idx] = f2bf(g * hbuf[idx]);            // r*h, bf16 for next GEMM
      }
    }
  }
}

// ---------------- step kernel 2: candidate + state update ----------------
// grid = 16 blocks, 64 cols per block.
__global__ void __launch_bounds__(256)
k_cand(int t, const float* __restrict__ x,
       const unsigned short* __restrict__ Whbf, const float* __restrict__ bh,
       const unsigned short* __restrict__ rhbf, float* __restrict__ hbuf,
       const float* __restrict__ zbuf, float* __restrict__ out,
       unsigned short* __restrict__ hbf) {
  __shared__ uint4 ldsA4[128 * 4];
  const int nblock = blockIdx.x * 64;
  const int wave   = threadIdx.x >> 5;
  const int lane   = threadIdx.x & 31;
  const int ncol   = nblock + (wave & 3) * 16;
  const int mbase  = (wave >> 2) * 64;

  v8f acc[4] = {};
  wave_gemm(acc, ldsA4, t, x, rhbf, Whbf, ncol, mbase, lane);

  const int lrow = lane & 15;
  const int lsel = (lane >> 4) & 1;
  const int j    = ncol + lrow;
  const float bj = bh[j];
  #pragma unroll
  for (int mi = 0; mi < 4; ++mi) {
    #pragma unroll
    for (int v = 0; v < 8; ++v) {
      const int    brow = mbase + mi * 16 + v + lsel * 8;
      const size_t idx  = (size_t)brow * HID + j;
      const float  ht = tanhf(acc[mi][v] + bj);
      const float  z  = zbuf[idx];
      const float  h  = hbuf[idx];
      const float  hn = (1.0f - z) * h + z * ht;
      out[((size_t)t * BATCH + brow) * HID + j] = hn;
      hbuf[idx] = hn;
      hbf[idx]  = f2bf(hn);
    }
  }
}

// ---------------- host-side launch ----------------
extern "C" void kernel_launch(void* const* d_in, const int* in_sizes, int n_in,
                              void* d_out, int out_size, void* d_ws, size_t ws_size,
                              hipStream_t stream) {
  (void)in_sizes; (void)n_in; (void)out_size; (void)ws_size;
  const float* x  = (const float*)d_in[0];
  const float* h0 = (const float*)d_in[1];
  const float* Wz = (const float*)d_in[2];
  const float* bz = (const float*)d_in[3];
  const float* Wr = (const float*)d_in[4];
  const float* br = (const float*)d_in[5];
  const float* Wh = (const float*)d_in[6];
  const float* bh = (const float*)d_in[7];
  float* out = (float*)d_out;

  // workspace layout (~11 MB total)
  unsigned short* Wzbf = (unsigned short*)d_ws;
  unsigned short* Wrbf = Wzbf + (size_t)HID * KTOT;
  unsigned short* Whbf = Wrbf + (size_t)HID * KTOT;
  unsigned short* hbf  = Whbf + (size_t)HID * KTOT;
  unsigned short* rhbf = hbf  + (size_t)BATCH * HID;
  float* hbuf = (float*)(rhbf + (size_t)BATCH * HID);
  float* zbuf = hbuf + (size_t)BATCH * HID;

  k_convert_w<<<2048, 256, 0, stream>>>(Wz, Wr, Wh, Wzbf, Wrbf, Whbf);
  k_init_h<<<(BATCH * HID) / 256, 256, 0, stream>>>(h0, hbuf, hbf);

  for (int t = 0; t < LSEQ; ++t) {
    k_gates<<<32, 256, 0, stream>>>(t, x, Wzbf, Wrbf, bz, br, hbf, hbuf, zbuf, rhbf);
    k_cand<<<16, 256, 0, stream>>>(t, x, Whbf, bh, rhbf, hbuf, zbuf, out, hbf);
  }
}
